// BasicGNN_23330262351896
// MI455X (gfx1250) — matrix operand order repaired and checked by reading, not compiled, not run on hardware
//
#include <hip/hip_runtime.h>

// CDNA5 / gfx1250: wave32, WMMA 16x16x32 f16->f32 accumulate.
typedef __attribute__((ext_vector_type(16))) _Float16 v16h;
typedef __attribute__((ext_vector_type(8)))  float    v8f;

// ---------------- degree / normalization ----------------

__global__ void k_deg_init(float* __restrict__ deg, int n) {
  int i = blockIdx.x * blockDim.x + threadIdx.x;
  if (i < n) deg[i] = 1.0f;  // self loop contributes 1 to every target degree
}

__global__ void k_deg_edges(const int* __restrict__ eidx, int E, float* __restrict__ deg) {
  int e = blockIdx.x * blockDim.x + threadIdx.x;
  if (e >= E) return;
  int c = __builtin_nontemporal_load(eidx + E + e);  // col = target
  atomicAdd(&deg[c], 1.0f);
}

__global__ void k_dinv(float* __restrict__ deg, int n) {
  int i = blockIdx.x * blockDim.x + threadIdx.x;
  if (i < n) deg[i] = rsqrtf(deg[i]);  // deg >= 1 always (self loop), no zero case
}

// ---------------- GEMM1: h1[n,16] = x[n,128] @ W1[128,16]  (WMMA f16) ----------------
// One wave per 16-row output tile. A layout (16-bit A 16x32): lane L<16 holds row M=L,
// halves {K..K+7, K+16..K+23}; lane L+16 holds {K+8..K+15, K+24..K+31}. B layout
// (16-bit B 32x16): lanes 0-15 column N=lane, K=0..7,16..23 across 8 VGPRs; lanes 16-31 K=8..15,24..31.
// C/D: VGPR v <-> M = v + 8*(lane>=16), N = lane&15.

__global__ void k_gemm1(const float* __restrict__ x, const float* __restrict__ W1,
                        float* __restrict__ h1, int n) {
  const int wave  = threadIdx.x >> 5;
  const int lane  = threadIdx.x & 31;
  const int tileM = blockIdx.x * (blockDim.x >> 5) + wave;
  if (tileM * 16 >= n) return;                 // wave-uniform exit
  const int m  = lane & 15;
  const int kh = lane >> 4;
  int row = tileM * 16 + m;
  if (row >= n) row = n - 1;                   // keep EXEC full for WMMA
  const float* xr = x + (long long)row * 128;

  v8f acc = {};
#pragma unroll
  for (int k0 = 0; k0 < 128; k0 += 32) {
    v16h a, b;
#pragma unroll
    for (int j = 0; j < 8; ++j) {
      a[j]     = (_Float16)__builtin_nontemporal_load(xr + k0 + kh * 8 + j);
      a[8 + j] = (_Float16)__builtin_nontemporal_load(xr + k0 + 16 + kh * 8 + j);
    }
#pragma unroll
    for (int j = 0; j < 16; ++j)
      b[j] = (_Float16)W1[(k0 + kh * 8 + (j & 7) + (j >> 3) * 16) * 16 + m];  // W1 row-major [K][N]
    acc = __builtin_amdgcn_wmma_f32_16x16x32_f16(false, a, false, b, (short)0, acc,
                                                 false, false);
  }
#pragma unroll
  for (int v = 0; v < 8; ++v) {
    int r = tileM * 16 + kh * 8 + v;
    if (r < n) h1[(long long)r * 16 + m] = acc[v];
  }
}

// ---------------- GEMM2: h2[n,40] = rel[n,16] @ W2[16,40]  (WMMA f16, K padded to 32) ----

__global__ void k_gemm2(const float* __restrict__ rel, const float* __restrict__ W2,
                        float* __restrict__ h2, int n) {
  const int wave = threadIdx.x >> 5;
  const int lane = threadIdx.x & 31;
  const int tile = blockIdx.x * (blockDim.x >> 5) + wave;
  const int mTiles = (n + 15) >> 4;
  if (tile >= mTiles * 3) return;              // 3 N-tiles cover 40 columns
  const int tileM = tile / 3;
  const int tileN = tile - tileM * 3;
  const int m  = lane & 15;
  const int kh = lane >> 4;
  int row = tileM * 16 + m;
  if (row >= n) row = n - 1;
  const int col = tileN * 16 + m;

  v16h a = {}, b = {};                          // zero-pad K in [16,32)
#pragma unroll
  for (int j = 0; j < 8; ++j)                   // valid K = kh*8 + j < 16
    a[j] = (_Float16)rel[(long long)row * 16 + kh * 8 + j];
  if (col < 40) {                               // B: lane half kh carries K=kh*8..kh*8+7
#pragma unroll
    for (int j = 0; j < 8; ++j)
      b[j] = (_Float16)W2[(kh * 8 + j) * 40 + col];  // W2 row-major [16][40]
  }
  v8f acc = {};
  acc = __builtin_amdgcn_wmma_f32_16x16x32_f16(false, a, false, b, (short)0, acc,
                                               false, false);
#pragma unroll
  for (int v = 0; v < 8; ++v) {
    int r = tileM * 16 + kh * 8 + v;
    if (r < n && col < 40) h2[(long long)r * 40 + col] = acc[v];
  }
}

// ---------------- elementwise: self-loop init (+ optional bias) ----------------

template <int F>
__global__ void k_selfloop(const float* __restrict__ h, const float* __restrict__ dinv,
                           const float* __restrict__ bias, float* __restrict__ out, int n) {
  long long t = (long long)blockIdx.x * blockDim.x + threadIdx.x;
  if (t >= (long long)n * F) return;
  int i = (int)(t / F);
  int f = (int)(t - (long long)i * F);
  float d = dinv[i];
  float v = h[t] * d * d;                       // self-loop norm = dinv[i]^2
  if (bias) v += bias[f];
  out[t] = v;
}

__global__ void k_relu_bias16(const float* __restrict__ agg, const float* __restrict__ b,
                              float* __restrict__ out, int n) {
  long long t = (long long)blockIdx.x * blockDim.x + threadIdx.x;
  if (t >= (long long)n * 16) return;
  int f = (int)(t & 15);
  out[t] = fmaxf(agg[t] + b[f], 0.0f);
}

// ---------------- per-edge gather/scale/scatter (L2-resident atomics) ----------------

template <int F>
__global__ void k_agg_edges(const int* __restrict__ eidx, int E,
                            const float* __restrict__ dinv,
                            const float* __restrict__ h, float* __restrict__ out) {
  int e = blockIdx.x * blockDim.x + threadIdx.x;
  if (e >= E) return;
  int r = __builtin_nontemporal_load(eidx + e);      // source
  int c = __builtin_nontemporal_load(eidx + E + e);  // target
  float w = dinv[r] * dinv[c];
  const float4* hr = (const float4*)(h + (long long)r * F);
  float* oc = out + (long long)c * F;
#pragma unroll
  for (int q = 0; q < F / 4; ++q) {
    float4 v = hr[q];
    atomicAdd(oc + 4 * q + 0, v.x * w);
    atomicAdd(oc + 4 * q + 1, v.y * w);
    atomicAdd(oc + 4 * q + 2, v.z * w);
    atomicAdd(oc + 4 * q + 3, v.w * w);
  }
}

// ---------------- launch ----------------

extern "C" void kernel_launch(void* const* d_in, const int* in_sizes, int n_in,
                              void* d_out, int out_size, void* d_ws, size_t ws_size,
                              hipStream_t stream) {
  const float* x  = (const float*)d_in[0];
  const int*   ei = (const int*)d_in[1];
  const float* W1 = (const float*)d_in[2];
  const float* b1 = (const float*)d_in[3];
  const float* W2 = (const float*)d_in[4];
  const float* b2 = (const float*)d_in[5];
  float* out = (float*)d_out;

  const int n = in_sizes[0] / 128;   // 100000
  const int E = in_sizes[1] / 2;     // 3200000

  // workspace layout (floats): dinv | h1[n*16] | agg1[n*16] | h2[n*40]  (~29 MB)
  float* ws   = (float*)d_ws;
  size_t oH1  = (size_t)((n + 127) & ~127);
  float* dinv = ws;
  float* h1   = ws + oH1;
  float* agg1 = h1 + (size_t)n * 16;
  float* h2   = agg1 + (size_t)n * 16;

  const int B   = 256;              // 8 waves per block on wave32
  const int wpb = B / 32;
  int gN = (n + B - 1) / B;
  int gE = (E + B - 1) / B;
  int mTiles = (n + 15) / 16;

  // normalization
  k_deg_init<<<gN, B, 0, stream>>>(dinv, n);
  k_deg_edges<<<gE, B, 0, stream>>>(ei, E, dinv);
  k_dinv<<<gN, B, 0, stream>>>(dinv, n);

  // layer 1
  k_gemm1<<<(mTiles + wpb - 1) / wpb, B, 0, stream>>>(x, W1, h1, n);
  long long n16 = (long long)n * 16;
  k_selfloop<16><<<(int)((n16 + B - 1) / B), B, 0, stream>>>(h1, dinv, nullptr, agg1, n);
  k_agg_edges<16><<<gE, B, 0, stream>>>(ei, E, dinv, h1, agg1);
  k_relu_bias16<<<(int)((n16 + B - 1) / B), B, 0, stream>>>(agg1, b1, h1, n);  // rel -> h1

  // layer 2
  int tiles2 = mTiles * 3;
  k_gemm2<<<(tiles2 + wpb - 1) / wpb, B, 0, stream>>>(h1, W2, h2, n);
  long long n40 = (long long)n * 40;
  k_selfloop<40><<<(int)((n40 + B - 1) / B), B, 0, stream>>>(h2, dinv, b2, out, n);
  k_agg_edges<40><<<gE, B, 0, stream>>>(ei, E, dinv, h2, out);
}
